// Attention_65137474011746
// MI455X (gfx1250) — matrix operand order repaired and checked by reading, not compiled, not run on hardware
//
#include <hip/hip_runtime.h>
#include <hip/hip_bf16.h>

#define BN  4
#define QN  2048
#define KNN 2048
#define CN  512
#define HN  8
#define DN  64
#define HDN 512

typedef __bf16 bf16;
typedef __attribute__((ext_vector_type(16))) __bf16 v16bf;
typedef __attribute__((ext_vector_type(8)))  float  v8f;
typedef unsigned int v4u __attribute__((ext_vector_type(4)));
typedef int v8i __attribute__((ext_vector_type(8)));
typedef int v4i __attribute__((ext_vector_type(4)));

#if __has_builtin(__builtin_amdgcn_tensor_load_to_lds)
#define HAVE_TDM 1
#else
#define HAVE_TDM 0
#endif

__device__ __forceinline__ bf16 f32_to_bf16(float f) {
  union { float f; unsigned u; } in; in.f = f;
  unsigned u = in.u;
  u += 0x7fffu + ((u >> 16) & 1u);           // round-to-nearest-even
  union { unsigned short s; bf16 b; } out; out.s = (unsigned short)(u >> 16);
  return out.b;
}

__device__ __forceinline__ v8f wmma_bf16(v16bf a, v16bf b, v8f c) {
  // D(f32 16x16) = A(bf16 16x32) * B(bf16 32x16) + C
  return __builtin_amdgcn_wmma_f32_16x16x32_bf16(
      false, a, false, b, (short)0, c, false, false);
}

// CDNA5 async cache->LDS copy, 16B per lane; tracked by ASYNCcnt.
__device__ __forceinline__ void async_load_b128(const bf16* gsrc, bf16* ldst) {
  const unsigned lds = (unsigned)(size_t)(void*)ldst;           // LDS offset
  const unsigned long long ga = (unsigned long long)(size_t)gsrc;
  asm volatile("global_load_async_to_lds_b128 %0, %1, off"
               :: "v"(lds), "v"(ga) : "memory");
}
__device__ __forceinline__ void wait_async0() {
  asm volatile("s_wait_asynccnt 0" ::: "memory");
}
__device__ __forceinline__ void wait_tensor0() {
#if __has_builtin(__builtin_amdgcn_s_wait_tensorcnt)
  __builtin_amdgcn_s_wait_tensorcnt(0);
#else
  asm volatile("s_wait_tensorcnt 0" ::: "memory");
#endif
}

// A-matrix fragment (16x32 bf16, row-major source, row stride ld).
// Lanes 0-15: rows 0-15, K=0..7 then 16..23; lanes 16-31: K=8..15 then 24..31.
__device__ __forceinline__ v16bf load_a_frag(const bf16* src, int ld) {
  const int lane = threadIdx.x & 31;
  const int m = lane & 15, h = lane >> 4;
  const bf16* row = src + (size_t)m * ld;
  v16bf a;
#pragma unroll
  for (int j = 0; j < 8; ++j) a[j] = row[h * 8 + j];
#pragma unroll
  for (int j = 0; j < 8; ++j) a[8 + j] = row[16 + h * 8 + j];
  return a;
}

// B-matrix fragment (32x16 bf16). Source holds B^T row-major: src[n*ld + k].
// Lanes 0-15: K=0..7 then 16..23; lanes 16-31: K=8..15 then 24..31, column n = lane&15.
__device__ __forceinline__ v16bf load_b_frag(const bf16* src, int ld) {
  const int lane = threadIdx.x & 31;
  const int n = lane & 15, kh = lane >> 4;
  const bf16* row = src + (size_t)n * ld + kh * 8;
  v16bf b;
#pragma unroll
  for (int t = 0; t < 16; ++t) b[t] = row[t < 8 ? t : t + 8];
  return b;
}

// ---------------------------------------------------------------- convert
__global__ void cvt_f32_to_bf16(const float* __restrict__ in,
                                bf16* __restrict__ out, int n) {
  for (int i = blockIdx.x * blockDim.x + threadIdx.x; i < n;
       i += gridDim.x * blockDim.x)
    out[i] = f32_to_bf16(in[i]);
}

// ---------------------------------------------------------------- GEMM
// Y[M,N] = alpha * X[M,K] * W[N,K]^T (+ bias[n]); stores bf16 and/or f32.
// Block: 256 thr = 8 waves; tile 128(M) x 128(N); wave tile 32x64; K-step 64.
// Double-buffered LDS, filled with global_load_async_to_lds_b128 (ASYNCcnt).
__global__ __launch_bounds__(256) void bf16_gemm_nt(
    const bf16* __restrict__ X, const bf16* __restrict__ W,
    int M, int N, int Kd, float alpha, const float* __restrict__ bias,
    bf16* __restrict__ Yb, float* __restrict__ Yf) {
  constexpr int LDT = 72;  // 64 + 8 pad (144B rows, 16B aligned)
  __shared__ alignas(16) bf16 sX[2][128][LDT];
  __shared__ alignas(16) bf16 sW[2][128][LDT];

  const int tid = threadIdx.x;
  const int w = tid >> 5, lane = tid & 31;
  const int hh = lane >> 4, ln = lane & 15;
  const int gm0 = blockIdx.y * 128, gn0 = blockIdx.x * 128;
  const int wm = (w >> 1) * 32, wn = (w & 1) * 64;

  v8f acc[2][4];
#pragma unroll
  for (int i = 0; i < 2; ++i)
#pragma unroll
    for (int j = 0; j < 4; ++j)
#pragma unroll
      for (int r = 0; r < 8; ++r) acc[i][j][r] = 0.f;

  auto stage = [&](int buf, int k0) {
#pragma unroll
    for (int it = 0; it < 4; ++it) {       // X tile 128x64: 1024 x b128
      const int idx = tid + it * 256;
      const int row = idx >> 3, c8 = idx & 7;
      async_load_b128(X + (size_t)(gm0 + row) * Kd + k0 + c8 * 8,
                      &sX[buf][row][c8 * 8]);
    }
#pragma unroll
    for (int it = 0; it < 4; ++it) {       // W tile 128x64
      const int idx = tid + it * 256;
      const int row = idx >> 3, c8 = idx & 7;
      async_load_b128(W + (size_t)(gn0 + row) * Kd + k0 + c8 * 8,
                      &sW[buf][row][c8 * 8]);
    }
  };

  stage(0, 0);
  int cur = 0;
  for (int k0 = 0; k0 < Kd; k0 += 64) {
    wait_async0();          // current buffer filled
    __syncthreads();        // ...and other buffer fully consumed by all waves
    if (k0 + 64 < Kd) stage(cur ^ 1, k0 + 64);  // overlap next fill w/ compute

#pragma unroll
    for (int ks = 0; ks < 64; ks += 32) {
      v16bf af[2], bfr[4];
#pragma unroll
      for (int i = 0; i < 2; ++i)
        af[i] = load_a_frag(&sX[cur][wm + i * 16][ks], LDT);
#pragma unroll
      for (int j = 0; j < 4; ++j)
        bfr[j] = load_b_frag(&sW[cur][wn + j * 16][ks], LDT);
#pragma unroll
      for (int i = 0; i < 2; ++i)
#pragma unroll
        for (int j = 0; j < 4; ++j)
          acc[i][j] = wmma_bf16(af[i], bfr[j], acc[i][j]);
    }
    cur ^= 1;
  }

#pragma unroll
  for (int i = 0; i < 2; ++i)
#pragma unroll
    for (int j = 0; j < 4; ++j)
#pragma unroll
      for (int r = 0; r < 8; ++r) {
        const int m = gm0 + wm + i * 16 + r + 8 * hh;
        const int n = gn0 + wn + j * 16 + ln;
        float v = acc[i][j][r] * alpha;
        if (bias) v += bias[n];
        const size_t o = (size_t)m * N + n;
        if (Yb) Yb[o] = f32_to_bf16(v);
        if (Yf) Yf[o] = v;
      }
}

// ---------------------------------------------------------------- flash attention
// Grid: (QN/128, BN*HN). 8 waves; each wave owns 16 query rows.
// K-tile via Tensor Data Mover (TDM) with padded-LDS layout; online softmax
// over 64-key chunks; S and P*V via bf16 WMMA, f32 accum.
__global__ __launch_bounds__(256) void flash_attn(
    const bf16* __restrict__ Qm, const bf16* __restrict__ Km,
    const bf16* __restrict__ Vm, const float* __restrict__ bmask,
    const float* __restrict__ bpair, float* __restrict__ O) {
  constexpr int LDK = 72;  // 64 + 8 pad (TDM: pad 4 dwords per 32 dwords)
  __shared__ alignas(16) bf16 sK[64][LDK];        // [key][d]
  __shared__ alignas(16) bf16 sVt[64][LDK];       // [d][key]  (transposed)
  __shared__ alignas(16) bf16 sP[8][16][LDK];     // per-wave P staging

  const int bh = blockIdx.y;
  const int b = bh / HN, h = bh % HN;
  const int tid = threadIdx.x;
  const int w = tid >> 5, lane = tid & 31;
  const int hh = lane >> 4, ln = lane & 15;
  const int q0 = blockIdx.x * 128 + w * 16;

  const bf16* qbase = Qm + ((size_t)b * QN + q0) * HDN + h * DN;
  v16bf qf[2];
  qf[0] = load_a_frag(qbase, HDN);       // d = 0..31
  qf[1] = load_a_frag(qbase + 32, HDN);  // d = 32..63

  float mrun[8], lrun[8];
  v8f ot[4];
#pragma unroll
  for (int r = 0; r < 8; ++r) { mrun[r] = -3.0e38f; lrun[r] = 0.f; }
#pragma unroll
  for (int dt = 0; dt < 4; ++dt)
#pragma unroll
    for (int r = 0; r < 8; ++r) ot[dt][r] = 0.f;

  for (int kc = 0; kc < KNN; kc += 64) {
    __syncthreads();
    const bf16* kb = Km + ((size_t)b * KNN + kc) * HDN + h * DN;
#if HAVE_TDM
    if (w == 0) {  // one wave drives the TDM for the whole workgroup
      const unsigned long long ga = (unsigned long long)(size_t)kb;
      const unsigned ldsoff = (unsigned)(size_t)(void*)&sK[0][0];
      v4u g0;
      g0[0] = 1u;                                   // count=1, no gather
      g0[1] = ldsoff;                               // lds_addr
      g0[2] = (unsigned)ga;                         // global_addr[31:0]
      g0[3] = (unsigned)((ga >> 32) & 0x1FFFFFFu)   // global_addr[56:32]
              | (2u << 30);                         // type=2 (image)
      const unsigned td0 = 1u << 20, td1 = 1u << 20;
      v8i g1;
      g1[0] = (int)((1u << 16)      // data_size = 2 bytes
                    | (1u << 20)    // pad_enable
                    | (4u << 22)    // pad_interval: every 32 dwords (128B)
                    | (3u << 25));  // pad_amount: 4 dwords (16B) -> LDK=72
      g1[1] = (int)((td0 & 0xFFFFu) << 16);
      g1[2] = (int)(((td0 >> 16) & 0xFFFFu) | ((td1 & 0xFFFFu) << 16));
      g1[3] = (int)(((td1 >> 16) & 0xFFFFu) | (64u << 16));  // tile_dim0=64
      g1[4] = 64;                                            // tile_dim1=64
      g1[5] = HDN;                                           // dim0 stride
      g1[6] = 0; g1[7] = 0;
      v4i g2 = {0, 0, 0, 0}, g3 = {0, 0, 0, 0};
      v8i g4 = {0, 0, 0, 0, 0, 0, 0, 0};
      __builtin_amdgcn_tensor_load_to_lds(g0, g1, g2, g3, g4, 0);
    }
#else
#pragma unroll
    for (int it = 0; it < 2; ++it) {     // K chunk 64x64: async fill
      const int idx = tid + it * 256;
      const int row = idx >> 3, c8 = idx & 7;
      async_load_b128(kb + (size_t)row * HDN + c8 * 8, &sK[row][c8 * 8]);
    }
#endif
    const bf16* vb = Vm + ((size_t)b * KNN + kc) * HDN + h * DN;
#pragma unroll
    for (int i = 0; i < 16; ++i) {       // V chunk, transposed into LDS
      const int idx = tid + i * 256;
      const int key = idx >> 6, dd = idx & 63;
      sVt[dd][key] = vb[(size_t)key * HDN + dd];
    }
    if (kc + 64 < KNN)                    // warm L2 for next bias_pair slab
      __builtin_prefetch(bpair + ((size_t)h * QN + q0) * KNN + kc + 64, 0, 1);
#if HAVE_TDM
    if (w == 0) wait_tensor0();
#else
    wait_async0();
#endif
    __syncthreads();

    // S = Q * K^T  (4 tiles of 16 keys)
    v8f st[4];
#pragma unroll
    for (int nt = 0; nt < 4; ++nt) {
      v8f a;
#pragma unroll
      for (int r = 0; r < 8; ++r) a[r] = 0.f;
      v16bf b0 = load_b_frag(&sK[nt * 16][0], LDK);   // d 0..31
      v16bf b1 = load_b_frag(&sK[nt * 16][32], LDK);  // d 32..63
      a = wmma_bf16(qf[0], b0, a);
      a = wmma_bf16(qf[1], b1, a);
      st[nt] = a;
    }

    // add bias_mask[b,k] + bias_pair[h,q,k]
#pragma unroll
    for (int nt = 0; nt < 4; ++nt) {
      const int kcol = kc + nt * 16 + ln;
      const float bm = bmask[(size_t)b * KNN + kcol];
      const float* bp = bpair + ((size_t)h * QN + q0 + 8 * hh) * KNN + kcol;
#pragma unroll
      for (int r = 0; r < 8; ++r) st[nt][r] += bm + bp[(size_t)r * KNN];
    }

    // online softmax (row = r + 8*hh, striped over 16 lanes per half)
    float rmax[8];
#pragma unroll
    for (int r = 0; r < 8; ++r)
      rmax[r] = fmaxf(fmaxf(st[0][r], st[1][r]), fmaxf(st[2][r], st[3][r]));
#pragma unroll
    for (int off = 8; off >= 1; off >>= 1)
#pragma unroll
      for (int r = 0; r < 8; ++r)
        rmax[r] = fmaxf(rmax[r], __shfl_xor(rmax[r], off, 16));

#pragma unroll
    for (int r = 0; r < 8; ++r) {
      const float mnew = fmaxf(mrun[r], rmax[r]);
      const float corr = __expf(mrun[r] - mnew);
      mrun[r] = mnew;
      float ps = 0.f;
#pragma unroll
      for (int nt = 0; nt < 4; ++nt) {
        const float p = __expf(st[nt][r] - mnew);
        sP[w][r + 8 * hh][nt * 16 + ln] = f32_to_bf16(p);
        ps += p;
      }
#pragma unroll
      for (int off = 8; off >= 1; off >>= 1) ps += __shfl_xor(ps, off, 16);
      lrun[r] = lrun[r] * corr + ps;
#pragma unroll
      for (int dt = 0; dt < 4; ++dt) ot[dt][r] *= corr;
    }

    // O += P * V   (P: 16x64 A-frags from per-wave LDS; V^T from sVt)
    v16bf pf0 = load_a_frag(&sP[w][0][0], LDK);   // keys 0..31
    v16bf pf1 = load_a_frag(&sP[w][0][32], LDK);  // keys 32..63
#pragma unroll
    for (int dt = 0; dt < 4; ++dt) {
      v16bf v0 = load_b_frag(&sVt[dt * 16][0], LDK);
      v16bf v1 = load_b_frag(&sVt[dt * 16][32], LDK);
      ot[dt] = wmma_bf16(pf0, v0, ot[dt]);
      ot[dt] = wmma_bf16(pf1, v1, ot[dt]);
    }
  }

  // normalize and store O as [B,Q,H*D] f32
#pragma unroll
  for (int r = 0; r < 8; ++r) {
    const float inv = 1.0f / lrun[r];
    float* orow = O + ((size_t)b * QN + q0 + r + 8 * hh) * HDN + h * DN;
#pragma unroll
    for (int dt = 0; dt < 4; ++dt) orow[dt * 16 + ln] = ot[dt][r] * inv;
  }
}

// ---------------------------------------------------------------- gating
__global__ void gate_mul(const float* __restrict__ o, const float* __restrict__ gl,
                         bf16* __restrict__ og, int n) {
  for (int i = blockIdx.x * blockDim.x + threadIdx.x; i < n;
       i += gridDim.x * blockDim.x) {
    const float g = 1.0f / (1.0f + __expf(-gl[i]));
    og[i] = f32_to_bf16(o[i] * g);
  }
}

// ---------------------------------------------------------------- launch
extern "C" void kernel_launch(void* const* d_in, const int* in_sizes, int n_in,
                              void* d_out, int out_size, void* d_ws, size_t ws_size,
                              hipStream_t stream) {
  (void)in_sizes; (void)n_in; (void)out_size; (void)ws_size;
  const float* q_x   = (const float*)d_in[0];
  const float* kv_x  = (const float*)d_in[1];
  const float* bmask = (const float*)d_in[2];
  const float* bpair = (const float*)d_in[3];
  const float* Wq    = (const float*)d_in[4];
  const float* Wk    = (const float*)d_in[5];
  const float* Wv    = (const float*)d_in[6];
  const float* Wo    = (const float*)d_in[7];
  const float* bo    = (const float*)d_in[8];
  const float* Wg    = (const float*)d_in[9];
  const float* bg    = (const float*)d_in[10];
  float* out = (float*)d_out;

  char* ws = (char*)d_ws;
  size_t off = 0;
  auto take = [&](size_t bytes) {
    char* p = ws + off;
    off = (off + bytes + 255) & ~(size_t)255;
    return p;
  };

  const size_t nQX = (size_t)BN * QN * CN;
  const size_t nKX = (size_t)BN * KNN * CN;
  const size_t nW  = (size_t)HDN * CN;
  const size_t nP  = (size_t)BN * QN * HDN;

  bf16* qx_b = (bf16*)take(nQX * 2);
  bf16* kx_b = (bf16*)take(nKX * 2);
  bf16* Wq_b = (bf16*)take(nW * 2);
  bf16* Wk_b = (bf16*)take(nW * 2);
  bf16* Wv_b = (bf16*)take(nW * 2);
  bf16* Wg_b = (bf16*)take(nW * 2);
  bf16* Wo_b = (bf16*)take(nW * 2);
  bf16* q_b  = (bf16*)take(nP * 2);
  bf16* k_b  = (bf16*)take(nP * 2);
  bf16* v_b  = (bf16*)take(nP * 2);
  float* gl   = (float*)take(nP * 4);
  float* oacc = (float*)take(nP * 4);
  bf16* og_b = (bf16*)take(nP * 2);

  const dim3 blk(256);
  cvt_f32_to_bf16<<<2048, blk, 0, stream>>>(q_x,  qx_b, (int)nQX);
  cvt_f32_to_bf16<<<2048, blk, 0, stream>>>(kv_x, kx_b, (int)nKX);
  cvt_f32_to_bf16<<<256, blk, 0, stream>>>(Wq, Wq_b, (int)nW);
  cvt_f32_to_bf16<<<256, blk, 0, stream>>>(Wk, Wk_b, (int)nW);
  cvt_f32_to_bf16<<<256, blk, 0, stream>>>(Wv, Wv_b, (int)nW);
  cvt_f32_to_bf16<<<256, blk, 0, stream>>>(Wg, Wg_b, (int)nW);
  cvt_f32_to_bf16<<<256, blk, 0, stream>>>(Wo, Wo_b, (int)nW);

  const dim3 pgrid(HDN / 128, (BN * QN) / 128);  // (4, 64)
  // q = (x Wq^T) / sqrt(D);  k = x Wk^T;  v = x Wv^T;  gate logits f32
  bf16_gemm_nt<<<pgrid, blk, 0, stream>>>(qx_b, Wq_b, BN * QN, HDN, CN, 0.125f,
                                          nullptr, q_b, nullptr);
  bf16_gemm_nt<<<pgrid, blk, 0, stream>>>(kx_b, Wk_b, BN * KNN, HDN, CN, 1.0f,
                                          nullptr, k_b, nullptr);
  bf16_gemm_nt<<<pgrid, blk, 0, stream>>>(kx_b, Wv_b, BN * KNN, HDN, CN, 1.0f,
                                          nullptr, v_b, nullptr);
  bf16_gemm_nt<<<pgrid, blk, 0, stream>>>(qx_b, Wg_b, BN * QN, HDN, CN, 1.0f,
                                          bg, nullptr, gl);

  const dim3 agrid(QN / 128, BN * HN);  // (16, 32)
  flash_attn<<<agrid, blk, 0, stream>>>(q_b, k_b, v_b, bmask, bpair, oacc);

  gate_mul<<<2048, blk, 0, stream>>>(oacc, gl, og_b, (int)nP);

  const dim3 ogrid(CN / 128, (BN * QN) / 128);
  bf16_gemm_nt<<<ogrid, blk, 0, stream>>>(og_b, Wo_b, BN * QN, CN, HDN, 1.0f,
                                          bo, nullptr, out);
}
